// VanillaRNN_76785425318196
// MI455X (gfx1250) — matrix operand order repaired and checked by reading, not compiled, not run on hardware
//
#include <hip/hip_runtime.h>
#include <cmath>

typedef _Float16 f16;
typedef __attribute__((ext_vector_type(16))) _Float16 v16h;
typedef __attribute__((ext_vector_type(8)))  _Float16 v8h;
typedef __attribute__((ext_vector_type(8)))  float    v8f;

#define BATCH 1024
#define SEQ   512
#define HID   512
#define OUT_N 10
#define KT    (HID / 32)    // 16 K-tiles of 32
#define MT    (HID / 16)    // 32 M-tiles
#define NT    (BATCH / 16)  // 64 N-tiles

// ---------------------------------------------------------------------------
// Pre-swizzle W_hh (fp32, [H,K] row-major) into this file's contiguous-k A-fragment
// layout for V_WMMA_F32_16X16X32_F16 (k = 16*hi + elem, not the ISA table order; B uses the same):
//   lane = (m%16) + 16*hi where hi = bit4 of k32
//   elem = k32 & 15
// Each lane's 16 f16 elems are stored contiguously (32B) -> b128 loads.
// ---------------------------------------------------------------------------
__global__ void prep_whh(const float* __restrict__ Whh, f16* __restrict__ Afrag) {
  int idx = blockIdx.x * blockDim.x + threadIdx.x;
  if (idx >= HID * HID) return;
  int m = idx / HID, k = idx % HID;
  int mt = m >> 4, lo = m & 15;
  int kt = k >> 5, k32 = k & 31;
  int hi = (k32 >> 4) & 1;
  int e  = k32 & 15;
  int lane = lo + (hi << 4);
  Afrag[(((mt * KT + kt) * 32 + lane) << 4) + e] = (f16)Whh[idx];
}

__global__ void zero_h(f16* __restrict__ Hbuf) {
  int idx = blockIdx.x * blockDim.x + threadIdx.x;
  if (idx < HID * BATCH) Hbuf[idx] = (f16)0.f;
}

// ---------------------------------------------------------------------------
// One recurrence step: Hout = tanh(W_hh @ Hin + W_hx * x[:,t] + b_h)
// Hin/Hout are in B-fragment layout: element (k,n) of the [K=H, N=B] matrix
// lives at tile (nt=n/16, kt=k/32), lane = (n%16) + 16*(k32>>4), elem = k32&15.
// Each wave computes a 32(M) x 32(N) patch = 2x2 WMMA tiles.
// Block = 256 threads = 8 waves (2 along M x 4 along N) -> 64(M) x 128(N).
// Grid = (8, 8). No divergence: EXEC is all-ones for every WMMA.
// ---------------------------------------------------------------------------
__global__ __launch_bounds__(256)
void rnn_step(const f16* __restrict__ Afrag,
              const f16* __restrict__ Hin,
              f16* __restrict__ Hout,
              const float* __restrict__ x,    // [B, T] row-major
              const float* __restrict__ Whx,  // [H]
              const float* __restrict__ bh,   // [H]
              int t) {
  const int lane = threadIdx.x & 31;
  const int wave = threadIdx.x >> 5;
  const int lo = lane & 15, hi = lane >> 4;
  const int wm = wave & 1, wn = wave >> 1;
  const int mt0 = blockIdx.x * 4 + wm * 2;  // 2 M-tiles per wave
  const int nt0 = blockIdx.y * 8 + wn * 2;  // 2 N-tiles per wave

  // C/D layout: VGPR v holds row m_local = v + 8*hi, column n_local = lo.
  // Initialize accumulators with the rank-1 + bias term in fp32.
  float xv[2];
#pragma unroll
  for (int j = 0; j < 2; ++j)
    xv[j] = x[((nt0 + j) * 16 + lo) * SEQ + t];

  v8f acc[2][2];
#pragma unroll
  for (int i = 0; i < 2; ++i) {
#pragma unroll
    for (int v = 0; v < 8; ++v) {
      int m = (mt0 + i) * 16 + v + 8 * hi;
      float w = Whx[m];
      float b = bh[m];
      acc[i][0][v] = fmaf(w, xv[0], b);
      acc[i][1][v] = fmaf(w, xv[1], b);
    }
  }

  const f16* aptr0 = Afrag + ((mt0 * KT) * 32 + lane) * 16;
  const f16* aptr1 = Afrag + (((mt0 + 1) * KT) * 32 + lane) * 16;
  const f16* bptr0 = Hin + ((nt0 * KT) * 32 + lane) * 16;
  const f16* bptr1 = Hin + (((nt0 + 1) * KT) * 32 + lane) * 16;

#pragma unroll 4
  for (int kt = 0; kt < KT; ++kt) {
    v16h a0 = *(const v16h*)(aptr0 + kt * 512);  // 32B/lane, 2x b128
    v16h a1 = *(const v16h*)(aptr1 + kt * 512);
    v16h b0 = *(const v16h*)(bptr0 + kt * 512);
    v16h b1 = *(const v16h*)(bptr1 + kt * 512);
    acc[0][0] = __builtin_amdgcn_wmma_f32_16x16x32_f16(false, a0, false, b0,
                                                       (short)0, acc[0][0], false, false);
    acc[0][1] = __builtin_amdgcn_wmma_f32_16x16x32_f16(false, a0, false, b1,
                                                       (short)0, acc[0][1], false, false);
    acc[1][0] = __builtin_amdgcn_wmma_f32_16x16x32_f16(false, a1, false, b0,
                                                       (short)0, acc[1][0], false, false);
    acc[1][1] = __builtin_amdgcn_wmma_f32_16x16x32_f16(false, a1, false, b1,
                                                       (short)0, acc[1][1], false, false);
  }

  // tanh in fp32, convert to f16, and scatter-store into B-fragment layout
  // for the next step. For output tile ht: kt_out = ht>>1, lane' = lo + 16*(ht&1),
  // elem = m_local = v + 8*hi -> 8 contiguous f16 = one 16B store per tile.
#pragma unroll
  for (int i = 0; i < 2; ++i) {
    const int ht = mt0 + i;
    const int ktO = ht >> 1;
    const int lanep = lo + ((ht & 1) << 4);
#pragma unroll
    for (int j = 0; j < 2; ++j) {
      const int bt = nt0 + j;
      v8h o;
#pragma unroll
      for (int v = 0; v < 8; ++v) o[v] = (f16)tanhf(acc[i][j][v]);
      *(v8h*)(Hout + ((bt * KT + ktO) * 32 + lanep) * 16 + 8 * hi) = o;
    }
  }
}

// ---------------------------------------------------------------------------
// Epilogue: y.T[b,o] = sum_k W_yh[o,k] * h_T[k,b] + b_y[o]  (10x1024, tiny)
// Reads h from the B-fragment layout.
// ---------------------------------------------------------------------------
__global__ void rnn_out(const f16* __restrict__ Hfrag,
                        const float* __restrict__ Wyh,  // [OUT, HID]
                        const float* __restrict__ by,   // [OUT]
                        float* __restrict__ yT) {       // [B, OUT]
  int idx = blockIdx.x * blockDim.x + threadIdx.x;
  if (idx >= BATCH * OUT_N) return;
  int b = idx / OUT_N, o = idx % OUT_N;
  int nt = b >> 4, nlo = b & 15;
  float s = by[o];
  for (int k = 0; k < HID; ++k) {
    int kt = k >> 5, k32 = k & 31;
    int lane = nlo + ((k32 >> 4) << 4);
    int e = k32 & 15;
    float hv = (float)Hfrag[(((nt * KT + kt) * 32 + lane) << 4) + e];
    s = fmaf(Wyh[o * HID + k], hv, s);
  }
  yT[idx] = s;
}

extern "C" void kernel_launch(void* const* d_in, const int* in_sizes, int n_in,
                              void* d_out, int out_size, void* d_ws, size_t ws_size,
                              hipStream_t stream) {
  (void)in_sizes; (void)n_in; (void)out_size; (void)ws_size;
  const float* x   = (const float*)d_in[0];  // [B, T]
  const float* Whx = (const float*)d_in[1];  // [H, 1]
  const float* Whh = (const float*)d_in[2];  // [H, H]
  const float* Wyh = (const float*)d_in[3];  // [O, H]
  const float* bh  = (const float*)d_in[4];  // [H, 1]
  const float* by  = (const float*)d_in[5];  // [O, 1]
  float* out = (float*)d_out;                // [B, O]

  char* ws = (char*)d_ws;
  f16* Afrag = (f16*)ws;                              // 512 KiB
  f16* h0    = (f16*)(ws + (size_t)HID * HID * 2);    // 1 MiB
  f16* h1    = (f16*)(ws + (size_t)HID * HID * 2 + (size_t)HID * BATCH * 2);

  prep_whh<<<(HID * HID + 255) / 256, 256, 0, stream>>>(Whh, Afrag);
  zero_h<<<(HID * BATCH + 255) / 256, 256, 0, stream>>>(h0);

  f16* hin = h0;
  f16* hout = h1;
  for (int t = 0; t < SEQ; ++t) {
    rnn_step<<<dim3(8, 8), 256, 0, stream>>>(Afrag, hin, hout, x, Whx, bh, t);
    f16* tmp = hin; hin = hout; hout = tmp;
  }
  rnn_out<<<(BATCH * OUT_N + 255) / 256, 256, 0, stream>>>(hin, Wyh, by, out);
}
